// ConLoss_89996744720824
// MI455X (gfx1250) — matrix operand order repaired and checked by reading, not compiled, not run on hardware
//
#include <hip/hip_runtime.h>
#include <hip/hip_bf16.h>

#define NROW 8192
#define DDIM 512
#define NBLK (NROW / 16)

typedef _Float16 v16h __attribute__((ext_vector_type(16)));
typedef _Float16 v8h  __attribute__((ext_vector_type(8)));
typedef float    v8f  __attribute__((ext_vector_type(8)));

union V16 { v16h v; v8h h[2]; };

// ---------------- kernel 1: fp32 -> f16 hi/lo split ----------------
__global__ void conloss_split(const float* __restrict__ f,
                              _Float16* __restrict__ hi,
                              _Float16* __restrict__ lo, int n) {
    int i = blockIdx.x * blockDim.x + threadIdx.x;
    if (i < n) {
        float x = f[i];
        _Float16 h = (_Float16)x;
        hi[i] = h;
        lo[i] = (_Float16)(x - (float)h);
    }
}

// ---------------- kernel 2: fused Z-tile WMMA + online LSE + pos stats ----------------
__global__ void __launch_bounds__(256)
conloss_main(const _Float16* __restrict__ Ahi,
             const _Float16* __restrict__ Alo,
             const int* __restrict__ labels,
             const float* __restrict__ weights,
             float* __restrict__ partial) {
    __shared__ float lds[8][16][4];   // [wave][row-in-tile][m,s,S,C]

    const int wave = threadIdx.x >> 5;
    const int lane = threadIdx.x & 31;
    const int hh   = lane >> 4;       // lane half (0/1)
    const int lr   = lane & 15;
    const int m0   = blockIdx.x * 16;
    const float invT = 10.0f;         // 1/0.1 exact

    // labels of the 8 M-rows this lane's C elements belong to
    int labm[8];
#pragma unroll
    for (int r = 0; r < 8; ++r) labm[r] = labels[m0 + hh * 8 + r];

    float mx[8], sm[8], Sp[8], Cn[8];
#pragma unroll
    for (int r = 0; r < 8; ++r) { mx[r] = -1e30f; sm[r] = 0.f; Sp[r] = 0.f; Cn[r] = 0.f; }

    const _Float16* pAh = Ahi + (size_t)(m0 + lr) * DDIM;  // A row = lane&15
    const _Float16* pAl = Alo + (size_t)(m0 + lr) * DDIM;

    for (int nb = wave * 32; nb < NROW; nb += 256) {
        const int n0 = nb, n1 = nb + 16;
        const int labn0 = labels[n0 + lr];
        const int labn1 = labels[n1 + lr];
        const _Float16* pB0h = Ahi + (size_t)(n0 + lr) * DDIM;
        const _Float16* pB0l = Alo + (size_t)(n0 + lr) * DDIM;
        const _Float16* pB1h = Ahi + (size_t)(n1 + lr) * DDIM;
        const _Float16* pB1l = Alo + (size_t)(n1 + lr) * DDIM;

        // prefetch next column-tile rows into cache (global_prefetch_b8)
        int npf = nb + 256;
        if (npf < NROW) {
            __builtin_prefetch(Ahi + (size_t)(npf + lr) * DDIM, 0, 3);
            __builtin_prefetch(Alo + (size_t)(npf + lr) * DDIM, 0, 3);
        }

        v8f c0 = {}; v8f c1 = {};
#pragma unroll 4
        for (int k = 0; k < DDIM; k += 32) {
            const int ka = k + hh * 16;   // A: lanes<16 -> K 0-15; lanes>=16 -> K 16-31 (contiguous)
            const int kb = k + hh * 16;   // B: lanes<16 -> K 0-15; lanes>=16 -> K 16-31 (contiguous)
            V16 ah, al, b0h, b0l, b1h, b1l;
            ah.h[0] = *(const v8h*)(pAh + ka);  ah.h[1] = *(const v8h*)(pAh + ka + 8);
            al.h[0] = *(const v8h*)(pAl + ka);  al.h[1] = *(const v8h*)(pAl + ka + 8);
            b0h.v = *(const v16h*)(pB0h + kb);
            b0l.v = *(const v16h*)(pB0l + kb);
            b1h.v = *(const v16h*)(pB1h + kb);
            b1l.v = *(const v16h*)(pB1l + kb);

            c0 = __builtin_amdgcn_wmma_f32_16x16x32_f16(false, ah.v, false, b0h.v, (short)0, c0, false, false);
            c1 = __builtin_amdgcn_wmma_f32_16x16x32_f16(false, ah.v, false, b1h.v, (short)0, c1, false, false);
            c0 = __builtin_amdgcn_wmma_f32_16x16x32_f16(false, ah.v, false, b0l.v, (short)0, c0, false, false);
            c1 = __builtin_amdgcn_wmma_f32_16x16x32_f16(false, ah.v, false, b1l.v, (short)0, c1, false, false);
            c0 = __builtin_amdgcn_wmma_f32_16x16x32_f16(false, al.v, false, b0h.v, (short)0, c0, false, false);
            c1 = __builtin_amdgcn_wmma_f32_16x16x32_f16(false, al.v, false, b1h.v, (short)0, c1, false, false);
        }

        // stats update: C element r -> row M = m0 + hh*8 + r, col N = n? + lr
#pragma unroll
        for (int r = 0; r < 8; ++r) {
            const int M = m0 + hh * 8 + r;
            {
                const int N = n0 + lr;
                if (N != M) {
                    float z = c0[r] * invT;
                    float nm = fmaxf(mx[r], z);
                    sm[r] = sm[r] * __expf(mx[r] - nm) + __expf(z - nm);
                    mx[r] = nm;
                    if (labn0 == labm[r]) { Sp[r] += z; Cn[r] += 1.0f; }
                }
            }
            {
                const int N = n1 + lr;
                if (N != M) {
                    float z = c1[r] * invT;
                    float nm = fmaxf(mx[r], z);
                    sm[r] = sm[r] * __expf(mx[r] - nm) + __expf(z - nm);
                    mx[r] = nm;
                    if (labn1 == labm[r]) { Sp[r] += z; Cn[r] += 1.0f; }
                }
            }
        }
    }

    // reduce across the 16 lanes that share each row (xor within lane halves)
#pragma unroll
    for (int r = 0; r < 8; ++r) {
        for (int off = 1; off < 16; off <<= 1) {
            float m2 = __shfl_xor(mx[r], off, 16);
            float s2 = __shfl_xor(sm[r], off, 16);
            float nm = fmaxf(mx[r], m2);
            sm[r] = sm[r] * __expf(mx[r] - nm) + s2 * __expf(m2 - nm);
            mx[r] = nm;
            Sp[r] += __shfl_xor(Sp[r], off, 16);
            Cn[r] += __shfl_xor(Cn[r], off, 16);
        }
    }
    if (lr == 0) {
#pragma unroll
        for (int r = 0; r < 8; ++r) {
            int row = hh * 8 + r;
            lds[wave][row][0] = mx[r];
            lds[wave][row][1] = sm[r];
            lds[wave][row][2] = Sp[r];
            lds[wave][row][3] = Cn[r];
        }
    }
    __syncthreads();

    // cross-wave merge + per-row result + deterministic per-block partial
    if (threadIdx.x < 16) {
        const int i = threadIdx.x;
        float M = -1e30f, S = 0.f, P = 0.f, C = 0.f;
#pragma unroll
        for (int w = 0; w < 8; ++w) {
            float m2 = lds[w][i][0], s2 = lds[w][i][1];
            float nm = fmaxf(M, m2);
            S = S * __expf(M - nm) + s2 * __expf(m2 - nm);
            M = nm;
            P += lds[w][i][2];
            C += lds[w][i][3];
        }
        float lse = M + __logf(S);
        float res = -weights[m0 + i] * (P - C * lse) / C;
        for (int off = 1; off < 16; off <<= 1) res += __shfl_xor(res, off, 16);
        if (i == 0) partial[blockIdx.x] = res;
    }
}

// ---------------- kernel 3: deterministic finalize ----------------
__global__ void conloss_finalize(const float* __restrict__ partial,
                                 const float* __restrict__ weights,
                                 float* __restrict__ out) {
    __shared__ float sA[256], sW[256];
    const int t = threadIdx.x;
    float a = 0.f, w = 0.f;
    for (int i = t; i < NBLK; i += 256) a += partial[i];
    for (int i = t; i < NROW; i += 256) w += weights[i];
    sA[t] = a; sW[t] = w;
    __syncthreads();
    for (int off = 128; off > 0; off >>= 1) {
        if (t < off) { sA[t] += sA[t + off]; sW[t] += sW[t + off]; }
        __syncthreads();
    }
    if (t == 0) out[0] = sA[0] / sW[0];
}

extern "C" void kernel_launch(void* const* d_in, const int* in_sizes, int n_in,
                              void* d_out, int out_size, void* d_ws, size_t ws_size,
                              hipStream_t stream) {
    const float* features = (const float*)d_in[0];
    const int*   labels   = (const int*)d_in[1];
    const float* weights  = (const float*)d_in[2];
    float* out = (float*)d_out;

    _Float16* Ahi = (_Float16*)d_ws;
    _Float16* Alo = Ahi + (size_t)NROW * DDIM;
    float* partial = (float*)(Alo + (size_t)NROW * DDIM);

    const int nelem = NROW * DDIM;
    conloss_split<<<(nelem + 255) / 256, 256, 0, stream>>>(features, Ahi, Alo, nelem);
    conloss_main<<<NBLK, 256, 0, stream>>>(Ahi, Alo, labels, weights, partial);
    conloss_finalize<<<1, 256, 0, stream>>>(partial, weights, out);
}
